// PyraformerEncoder_31937376813471
// MI455X (gfx1250) — matrix operand order repaired and checked by reading, not compiled, not run on hardware
//
#include <hip/hip_runtime.h>
#include <hip/hip_bf16.h>
#include <cstdint>
#include <cstddef>

// ---------------------------------------------------------------------------
// Types / WMMA fragment helpers (gfx1250, wave32, V_WMMA_F32_16X16X32_F16)
// ---------------------------------------------------------------------------
typedef __attribute__((ext_vector_type(16))) _Float16 v16h;
typedef __attribute__((ext_vector_type(8)))  float    v8f;

union FragH { v16h v; uint32_t u[8]; };
union AccF  { v8f  v; float    f[8]; };

// A-matrix 16x32 f16 layout (ISA 7.12.2): lanes 0-15 M=lane, K pairs:
//   vgpr 0-3: K = hi*8 + {0..7}; vgpr 4-7: K = 16 + hi*8 + {0..7}
__device__ __forceinline__ int koffA(int i, int hi) {
  return (i < 4) ? (hi * 8 + 2 * i) : (16 + hi * 8 + 2 * (i - 4));
}
// B-matrix 32x16 f16 layout: lane = column N, same K order as A:
//   vgpr 0-3: K = hi*8 + {0..7}; vgpr 4-7: K = 16 + hi*8 + {0..7}
__device__ __forceinline__ int koffB(int i, int hi) {
  return (i < 4) ? (hi * 8 + 2 * i) : (16 + hi * 8 + 2 * (i - 4));
}

__host__ __forceinline__ int cdiv(int a, int b) { return (a + b - 1) / b; }

// ---------------------------------------------------------------------------
// WMMA GEMM, 16x16 tile per wave (used for narrow N=32 case).
// C[M,N] = act(A[M,K] * Bt[N,K]^T + bias)
// ---------------------------------------------------------------------------
__global__ void gemm_wmma(const _Float16* __restrict__ A, int lda,
                          const _Float16* __restrict__ Bt, int ldb,
                          const float* __restrict__ bias,
                          float* __restrict__ Cf, _Float16* __restrict__ Ch,
                          int M, int N, int K, int relu) {
  const int lane = threadIdx.x & 31;
  const int wave = threadIdx.x >> 5;
  const int tiles_n = N >> 4;
  const int tiles_m = M >> 4;
  const int tile = blockIdx.x * (blockDim.x >> 5) + wave;
  if (tile >= tiles_m * tiles_n) return;     // whole-wave exit, EXEC uniform
  const int tm = tile / tiles_n;
  const int tn = tile % tiles_n;
  const int lo = lane & 15;
  const int hi = lane >> 4;

  const _Float16* Arow = A + (size_t)(tm * 16 + lo) * lda;
  const _Float16* Brow = Bt + (size_t)(tn * 16 + lo) * ldb;

  AccF acc;
#pragma unroll
  for (int r = 0; r < 8; ++r) acc.f[r] = 0.0f;

  for (int kc = 0; kc < K; kc += 32) {
    FragH a, b;
#pragma unroll
    for (int i = 0; i < 8; ++i) {
      a.u[i] = *(const uint32_t*)(Arow + kc + koffA(i, hi));
      b.u[i] = *(const uint32_t*)(Brow + kc + koffB(i, hi));
    }
    acc.v = __builtin_amdgcn_wmma_f32_16x16x32_f16(
        false, a.v, false, b.v, (short)0, acc.v, false, false);
  }

  const float bn = bias ? bias[tn * 16 + lo] : 0.0f;
#pragma unroll
  for (int r = 0; r < 8; ++r) {
    const int row = tm * 16 + hi * 8 + r;
    float val = acc.f[r] + bn;
    if (relu) val = val > 0.0f ? val : 0.0f;
    const size_t idx = (size_t)row * N + tn * 16 + lo;
    if (Cf) Cf[idx] = val;
    if (Ch) Ch[idx] = (_Float16)val;
  }
}

// ---------------------------------------------------------------------------
// WMMA GEMM, 16(M)x64(N) register block per wave: one A fragment feeds 4
// WMMAs (4x matrix-op density per A load, 4x register reuse of the B strip).
// Requires N % 64 == 0.
// ---------------------------------------------------------------------------
__global__ void gemm_wmma4(const _Float16* __restrict__ A, int lda,
                           const _Float16* __restrict__ Bt, int ldb,
                           const float* __restrict__ bias,
                           float* __restrict__ Cf, _Float16* __restrict__ Ch,
                           int M, int N, int K, int relu) {
  const int lane = threadIdx.x & 31;
  const int wave = threadIdx.x >> 5;
  const int tiles_n = N >> 6;                 // 64-wide N blocks
  const int tiles_m = M >> 4;
  const int tile = blockIdx.x * (blockDim.x >> 5) + wave;
  if (tile >= tiles_m * tiles_n) return;      // whole-wave exit, EXEC uniform
  const int tm = tile / tiles_n;
  const int tn = tile % tiles_n;
  const int lo = lane & 15;
  const int hi = lane >> 4;

  const _Float16* Arow = A + (size_t)(tm * 16 + lo) * lda;
  const _Float16* Brow0 = Bt + (size_t)(tn * 64 + lo) * ldb;
  const _Float16* Brow1 = Brow0 + (size_t)16 * ldb;
  const _Float16* Brow2 = Brow0 + (size_t)32 * ldb;
  const _Float16* Brow3 = Brow0 + (size_t)48 * ldb;

  AccF acc0, acc1, acc2, acc3;
#pragma unroll
  for (int r = 0; r < 8; ++r) { acc0.f[r] = 0.0f; acc1.f[r] = 0.0f; acc2.f[r] = 0.0f; acc3.f[r] = 0.0f; }

  for (int kc = 0; kc < K; kc += 32) {
    FragH a, b0, b1, b2, b3;
#pragma unroll
    for (int i = 0; i < 8; ++i) {
      a.u[i]  = *(const uint32_t*)(Arow  + kc + koffA(i, hi));
      const int kb = kc + koffB(i, hi);
      b0.u[i] = *(const uint32_t*)(Brow0 + kb);
      b1.u[i] = *(const uint32_t*)(Brow1 + kb);
      b2.u[i] = *(const uint32_t*)(Brow2 + kb);
      b3.u[i] = *(const uint32_t*)(Brow3 + kb);
    }
    acc0.v = __builtin_amdgcn_wmma_f32_16x16x32_f16(false, a.v, false, b0.v, (short)0, acc0.v, false, false);
    acc1.v = __builtin_amdgcn_wmma_f32_16x16x32_f16(false, a.v, false, b1.v, (short)0, acc1.v, false, false);
    acc2.v = __builtin_amdgcn_wmma_f32_16x16x32_f16(false, a.v, false, b2.v, (short)0, acc2.v, false, false);
    acc3.v = __builtin_amdgcn_wmma_f32_16x16x32_f16(false, a.v, false, b3.v, (short)0, acc3.v, false, false);
  }

#pragma unroll
  for (int j = 0; j < 4; ++j) {
    const AccF* accs[4] = { &acc0, &acc1, &acc2, &acc3 };
    const int col = tn * 64 + j * 16 + lo;
    const float bn = bias ? bias[col] : 0.0f;
#pragma unroll
    for (int r = 0; r < 8; ++r) {
      const int row = tm * 16 + hi * 8 + r;
      float val = accs[j]->f[r] + bn;
      if (relu) val = val > 0.0f ? val : 0.0f;
      const size_t idx = (size_t)row * N + col;
      if (Cf) Cf[idx] = val;
      if (Ch) Ch[idx] = (_Float16)val;
    }
  }
}

// ---------------------------------------------------------------------------
// Fused masked attention with online softmax.  One wave handles 16 query rows
// of one (batch, head).  q,k: [B*S,128] f16 (head-major cols); vt: [B,H,32,S]
// f16; mask: [S,S] bytes (1 = masked); o: [B*S,128] f16.
// ---------------------------------------------------------------------------
#define S_TOT 2720
#define NEGV  (-1.0e9f)

__global__ void attn_kernel(const _Float16* __restrict__ q,
                            const _Float16* __restrict__ k,
                            const _Float16* __restrict__ vt,
                            const unsigned char* __restrict__ mask,
                            _Float16* __restrict__ o) {
  __shared__ _Float16 Pl[16][40];   // P tile staging (C-layout -> A-layout)

  const int lane = threadIdx.x & 31;
  const int bid  = blockIdx.x;              // b*4*170 + h*170 + qt
  const int qt   = bid % 170;
  const int bh   = bid / 170;
  const int h    = bh % 4;
  const int b    = bh / 4;
  const int lo   = lane & 15;
  const int hi   = lane >> 4;
  const float inv_temp = 0.17677669529663687f;  // 1/sqrt(32)

  // Q fragment (A layout), K = full d_k = 32
  FragH qa;
  const _Float16* qrow = q + ((size_t)(b * S_TOT + qt * 16 + lo)) * 128 + h * 32;
#pragma unroll
  for (int i = 0; i < 8; ++i)
    qa.u[i] = *(const uint32_t*)(qrow + koffA(i, hi));

  const _Float16* kcol = k + ((size_t)(b * S_TOT + lo)) * 128 + h * 32;
  const _Float16* vrow0 = vt + ((size_t)((b * 4 + h) * 32 + lo)) * S_TOT;
  const _Float16* vrow1 = vt + ((size_t)((b * 4 + h) * 32 + 16 + lo)) * S_TOT;

  float m_r[8], l_r[8];
  AccF o0, o1;
#pragma unroll
  for (int r = 0; r < 8; ++r) { m_r[r] = -3.0e38f; l_r[r] = 0.0f; o0.f[r] = 0.0f; o1.f[r] = 0.0f; }

  for (int kc = 0; kc < S_TOT; kc += 32) {
    // ---- prefetch next chunk's K rows / V strip (global_prefetch_b8)
    if (kc + 32 < S_TOT) {
      __builtin_prefetch(kcol + (size_t)(kc + 32) * 128, 0, 1);
      __builtin_prefetch(kcol + (size_t)(kc + 48) * 128, 0, 1);
      __builtin_prefetch(vrow0 + kc + 32, 0, 1);
      __builtin_prefetch(vrow1 + kc + 32, 0, 1);
    }

    // ---- scores: two 16-key tiles via WMMA  (B frag = K rows, d contiguous)
    FragH kb0, kb1;
    const _Float16* kr0 = kcol + (size_t)kc * 128;
    const _Float16* kr1 = kr0 + 16 * 128;
#pragma unroll
    for (int i = 0; i < 8; ++i) {
      kb0.u[i] = *(const uint32_t*)(kr0 + koffB(i, hi));
      kb1.u[i] = *(const uint32_t*)(kr1 + koffB(i, hi));
    }
    AccF z;
#pragma unroll
    for (int r = 0; r < 8; ++r) z.f[r] = 0.0f;
    AccF s0, s1;
    s0.v = __builtin_amdgcn_wmma_f32_16x16x32_f16(false, qa.v, false, kb0.v, (short)0, z.v, false, false);
    s1.v = __builtin_amdgcn_wmma_f32_16x16x32_f16(false, qa.v, false, kb1.v, (short)0, z.v, false, false);

    __syncthreads();
    // ---- mask, scale, online softmax update; stage P into LDS
#pragma unroll
    for (int r = 0; r < 8; ++r) {
      const int row = qt * 16 + hi * 8 + r;
      const unsigned char* mr = mask + (size_t)row * S_TOT + kc;
      float a0 = s0.f[r] * inv_temp; if (mr[lo])      a0 = NEGV;
      float a1 = s1.f[r] * inv_temp; if (mr[16 + lo]) a1 = NEGV;
      float mx = fmaxf(a0, a1);
      mx = fmaxf(mx, __shfl_xor(mx, 1, 32));
      mx = fmaxf(mx, __shfl_xor(mx, 2, 32));
      mx = fmaxf(mx, __shfl_xor(mx, 4, 32));
      mx = fmaxf(mx, __shfl_xor(mx, 8, 32));
      const float mnew  = fmaxf(m_r[r], mx);
      const float scale = __expf(m_r[r] - mnew);
      const float p0 = __expf(a0 - mnew);
      const float p1 = __expf(a1 - mnew);
      float ps = p0 + p1;
      ps += __shfl_xor(ps, 1, 32);
      ps += __shfl_xor(ps, 2, 32);
      ps += __shfl_xor(ps, 4, 32);
      ps += __shfl_xor(ps, 8, 32);
      l_r[r] = l_r[r] * scale + ps;
      m_r[r] = mnew;
      o0.f[r] *= scale;
      o1.f[r] *= scale;
      Pl[hi * 8 + r][lo]      = (_Float16)p0;
      Pl[hi * 8 + r][16 + lo] = (_Float16)p1;
    }
    __syncthreads();

    // ---- repack P (A layout) from LDS
    FragH pa;
#pragma unroll
    for (int i = 0; i < 8; ++i)
      pa.u[i] = *(const uint32_t*)(&Pl[lo][koffA(i, hi)]);

    // ---- V fragments from Vt[b][h][d][s]  (s contiguous)
    FragH vb0, vb1;
#pragma unroll
    for (int i = 0; i < 8; ++i) {
      const int kb = kc + koffB(i, hi);
      vb0.u[i] = *(const uint32_t*)(vrow0 + kb);
      vb1.u[i] = *(const uint32_t*)(vrow1 + kb);
    }
    o0.v = __builtin_amdgcn_wmma_f32_16x16x32_f16(false, pa.v, false, vb0.v, (short)0, o0.v, false, false);
    o1.v = __builtin_amdgcn_wmma_f32_16x16x32_f16(false, pa.v, false, vb1.v, (short)0, o1.v, false, false);
  }

  // ---- finalize: O = acc / l
#pragma unroll
  for (int r = 0; r < 8; ++r) {
    const int row = qt * 16 + hi * 8 + r;
    const float inv = 1.0f / l_r[r];
    const size_t base = ((size_t)(b * S_TOT + row)) * 128 + h * 32;
    o[base + lo]      = (_Float16)(o0.f[r] * inv);
    o[base + 16 + lo] = (_Float16)(o1.f[r] * inv);
  }
}

// ---------------------------------------------------------------------------
// LayerNorm over D=128 rows: out = LN(x (+res)) * g + b ; one wave per row
// ---------------------------------------------------------------------------
__global__ void ln_kernel(const float* __restrict__ x, const float* __restrict__ res,
                          const float* __restrict__ g, const float* __restrict__ be,
                          float eps, float* __restrict__ out, _Float16* __restrict__ outh) {
  const int row  = blockIdx.x;
  const int lane = threadIdx.x;   // 32 threads
  const size_t base = (size_t)row * 128;
  float v[4];
#pragma unroll
  for (int j = 0; j < 4; ++j) {
    float t = x[base + lane + 32 * j];
    if (res) t += res[base + lane + 32 * j];
    v[j] = t;
  }
  float s = v[0] + v[1] + v[2] + v[3];
#pragma unroll
  for (int off = 16; off; off >>= 1) s += __shfl_xor(s, off, 32);
  const float mu = s * (1.0f / 128.0f);
  float q = 0.0f;
#pragma unroll
  for (int j = 0; j < 4; ++j) { const float d = v[j] - mu; q += d * d; }
#pragma unroll
  for (int off = 16; off; off >>= 1) q += __shfl_xor(q, off, 32);
  const float rstd = rsqrtf(q * (1.0f / 128.0f) + eps);
#pragma unroll
  for (int j = 0; j < 4; ++j) {
    const int c = lane + 32 * j;
    const float y = (v[j] - mu) * rstd * g[c] + be[c];
    out[base + c]  = y;
    outh[base + c] = (_Float16)y;
  }
}

// ---------------------------------------------------------------------------
// Conv1d (k=4, stride=4, VALID) + BatchNorm(inference) + ELU.
// Input indexed as in[b*inB + ch*inC + t*inT]; out layout [b][32][To].
// ---------------------------------------------------------------------------
__global__ void conv_bn_elu(const float* __restrict__ in, int inB, int inC, int inT,
                            const float* __restrict__ w, const float* __restrict__ cb,
                            const float* __restrict__ g, const float* __restrict__ be,
                            const float* __restrict__ mn, const float* __restrict__ vr,
                            float* __restrict__ out, int To) {
  const int idx = blockIdx.x * blockDim.x + threadIdx.x;
  const int total = 2 * 32 * To;
  if (idx >= total) return;
  const int p = idx % To;
  const int o = (idx / To) % 32;
  const int b = idx / (To * 32);
  float acc = cb[o];
  for (int i = 0; i < 32; ++i) {
#pragma unroll
    for (int t = 0; t < 4; ++t)
      acc += w[(o * 32 + i) * 4 + t] * in[(size_t)b * inB + (size_t)i * inC + (size_t)(4 * p + t) * inT];
  }
  acc = (acc - mn[o]) * rsqrtf(vr[o] + 1e-5f) * g[o] + be[o];
  out[(size_t)(b * 32 + o) * To + p] = acc > 0.0f ? acc : (__expf(acc) - 1.0f);
}

// ---------------------------------------------------------------------------
// Small elementwise / layout kernels
// ---------------------------------------------------------------------------
__global__ void cvt16(const float* __restrict__ s, _Float16* __restrict__ d, int n) {
  const int i = blockIdx.x * blockDim.x + threadIdx.x;
  if (i < n) d[i] = (_Float16)s[i];
}

// src[K][N] f32 -> dst[N][K] f16
__global__ void transpose16(const float* __restrict__ src, _Float16* __restrict__ dst,
                            int K, int N) {
  const int i = blockIdx.x * blockDim.x + threadIdx.x;
  if (i >= K * N) return;
  const int kk = i / N;
  const int nn = i % N;
  dst[(size_t)nn * K + kk] = (_Float16)src[(size_t)kk * N + nn];
}

// v[(b*S+s)*128 + h*32+d] -> vt[((b*4+h)*32+d)*S + s]
__global__ void vtrans(const _Float16* __restrict__ v, _Float16* __restrict__ vt) {
  const int idx = blockIdx.x * blockDim.x + threadIdx.x;
  if (idx >= 2 * S_TOT * 128) return;
  const int c = idx % 128;
  const int s = (idx / 128) % S_TOT;
  const int b = idx / (128 * S_TOT);
  const int h = c >> 5;
  const int d = c & 31;
  vt[((size_t)((b * 4 + h) * 32 + d)) * S_TOT + s] = v[idx];
}

// concat conv pyramid outputs -> pooled f16 [B*672][32]
__global__ void pool_assemble(const float* __restrict__ c1, const float* __restrict__ c2,
                              const float* __restrict__ c3, _Float16* __restrict__ pooled) {
  const int idx = blockIdx.x * blockDim.x + threadIdx.x;
  if (idx >= 2 * 672 * 32) return;
  const int c = idx % 32;
  const int p = (idx / 32) % 672;
  const int b = idx / (32 * 672);
  float v;
  if (p < 512)       v = c1[(size_t)(b * 32 + c) * 512 + p];
  else if (p < 640)  v = c2[(size_t)(b * 32 + c) * 128 + (p - 512)];
  else               v = c3[(size_t)(b * 32 + c) * 32 + (p - 640)];
  pooled[idx] = (_Float16)v;
}

// h[b][s][c] = s<2048 ? x[b][s][c] : up[b][s-2048][c]
__global__ void h_assemble(const float* __restrict__ x, const float* __restrict__ up,
                           float* __restrict__ h) {
  const int idx = blockIdx.x * blockDim.x + threadIdx.x;
  if (idx >= 2 * S_TOT * 128) return;
  const int c = idx % 128;
  const int s = (idx / 128) % S_TOT;
  const int b = idx / (128 * S_TOT);
  h[idx] = (s < 2048) ? x[((size_t)b * 2048 + s) * 128 + c]
                      : up[((size_t)b * 672 + (s - 2048)) * 128 + c];
}

// out[b][i][j*128+c] = h[b][indexes[i][j]][c]
__global__ void gather_out(const float* __restrict__ h, const int* __restrict__ indexes,
                           float* __restrict__ out) {
  const int idx = blockIdx.x * blockDim.x + threadIdx.x;
  if (idx >= 2 * 2048 * 4 * 128) return;
  const int c = idx % 128;
  const int j = (idx / 128) % 4;
  const int i = (idx / (128 * 4)) % 2048;
  const int b = idx / (128 * 4 * 2048);
  const int s = indexes[i * 4 + j];
  out[idx] = h[((size_t)b * S_TOT + s) * 128 + c];
}

// ---------------------------------------------------------------------------
// Host orchestration
// ---------------------------------------------------------------------------
extern "C" void kernel_launch(void* const* d_in, const int* in_sizes, int n_in,
                              void* d_out, int out_size, void* d_ws, size_t ws_size,
                              hipStream_t stream) {
  (void)in_sizes; (void)n_in; (void)out_size; (void)ws_size;
  const float* x       = (const float*)d_in[0];
  const float* w_down  = (const float*)d_in[1];
  const float* b_down  = (const float*)d_in[2];
  const float* conv_w  = (const float*)d_in[3];
  const float* conv_b  = (const float*)d_in[4];
  const float* bn_g    = (const float*)d_in[5];
  const float* bn_b    = (const float*)d_in[6];
  const float* bn_m    = (const float*)d_in[7];
  const float* bn_v    = (const float*)d_in[8];
  const float* w_up    = (const float*)d_in[9];
  const float* b_up    = (const float*)d_in[10];
  const float* ln0_g   = (const float*)d_in[11];
  const float* ln0_b   = (const float*)d_in[12];
  const float* wq      = (const float*)d_in[13];
  const float* wk      = (const float*)d_in[14];
  const float* wv      = (const float*)d_in[15];
  const float* wo      = (const float*)d_in[16];
  const float* ln1_g   = (const float*)d_in[17];
  const float* ln1_b   = (const float*)d_in[18];
  const float* ffn_w1  = (const float*)d_in[19];
  const float* ffn_b1  = (const float*)d_in[20];
  const float* ffn_w2  = (const float*)d_in[21];
  const float* ffn_b2  = (const float*)d_in[22];
  const float* ln2_g   = (const float*)d_in[23];
  const float* ln2_b   = (const float*)d_in[24];
  const unsigned char* mask = (const unsigned char*)d_in[25];
  const int* indexes   = (const int*)d_in[26];
  float* out = (float*)d_out;

  const int ROWS = 2 * S_TOT;     // 5440
  char* base = (char*)d_ws;
  size_t off = 0;
  auto alloc = [&](size_t bytes) -> char* {
    char* p = base + off;
    off = (off + bytes + 255) & ~(size_t)255;
    return p;
  };

  float*    hf     = (float*)   alloc((size_t)ROWS * 128 * 4);
  _Float16* hf16   = (_Float16*)alloc((size_t)ROWS * 128 * 2);
  float*    tmp    = (float*)   alloc((size_t)ROWS * 128 * 4);
  _Float16* xf16   = (_Float16*)alloc((size_t)4096 * 128 * 2);
  float*    t0     = (float*)   alloc((size_t)4096 * 32 * 4);
  float*    c1     = (float*)   alloc((size_t)2 * 32 * 512 * 4);
  float*    c2     = (float*)   alloc((size_t)2 * 32 * 128 * 4);
  float*    c3     = (float*)   alloc((size_t)2 * 32 * 32 * 4);
  _Float16* pooled = (_Float16*)alloc((size_t)1344 * 32 * 2);
  float*    tmpU   = (float*)   alloc((size_t)1344 * 128 * 4);
  _Float16* qf16   = (_Float16*)alloc((size_t)ROWS * 128 * 2);
  _Float16* kf16   = (_Float16*)alloc((size_t)ROWS * 128 * 2);
  _Float16* vf16   = (_Float16*)alloc((size_t)ROWS * 128 * 2);
  _Float16* vtf16  = (_Float16*)alloc((size_t)ROWS * 128 * 2);
  _Float16* of16   = (_Float16*)alloc((size_t)ROWS * 128 * 2);
  _Float16* f1f16  = (_Float16*)alloc((size_t)ROWS * 512 * 2);
  _Float16* wdt    = (_Float16*)alloc((size_t)32 * 128 * 2);
  _Float16* wupt   = (_Float16*)alloc((size_t)128 * 32 * 2);
  _Float16* wqt    = (_Float16*)alloc((size_t)128 * 128 * 2);
  _Float16* wkt    = (_Float16*)alloc((size_t)128 * 128 * 2);
  _Float16* wvt    = (_Float16*)alloc((size_t)128 * 128 * 2);
  _Float16* wot    = (_Float16*)alloc((size_t)128 * 128 * 2);
  _Float16* w1t    = (_Float16*)alloc((size_t)512 * 128 * 2);
  _Float16* w2t    = (_Float16*)alloc((size_t)128 * 512 * 2);

  // ---- bottleneck down projection: t0 = x @ w_down + b_down (N=32 -> 1-tile)
  cvt16<<<cdiv(4096 * 128, 256), 256, 0, stream>>>(x, xf16, 4096 * 128);
  transpose16<<<cdiv(128 * 32, 256), 256, 0, stream>>>(w_down, wdt, 128, 32);
  gemm_wmma<<<cdiv((4096 / 16) * (32 / 16), 8), 256, 0, stream>>>(
      xf16, 128, wdt, 128, b_down, t0, (_Float16*)nullptr, 4096, 32, 128, 0);

  // ---- conv pyramid (k=4, s=4) + BN + ELU
  conv_bn_elu<<<cdiv(2 * 32 * 512, 256), 256, 0, stream>>>(
      t0, 2048 * 32, 1, 32, conv_w, conv_b, bn_g, bn_b, bn_m, bn_v, c1, 512);
  conv_bn_elu<<<cdiv(2 * 32 * 128, 256), 256, 0, stream>>>(
      c1, 32 * 512, 512, 1, conv_w + 4096, conv_b + 32, bn_g + 32, bn_b + 32,
      bn_m + 32, bn_v + 32, c2, 128);
  conv_bn_elu<<<cdiv(2 * 32 * 32, 256), 256, 0, stream>>>(
      c2, 32 * 128, 128, 1, conv_w + 8192, conv_b + 64, bn_g + 64, bn_b + 64,
      bn_m + 64, bn_v + 64, c3, 32);

  // ---- pooled -> up projection -> assemble h -> LN0
  pool_assemble<<<cdiv(2 * 672 * 32, 256), 256, 0, stream>>>(c1, c2, c3, pooled);
  transpose16<<<cdiv(32 * 128, 256), 256, 0, stream>>>(w_up, wupt, 32, 128);
  gemm_wmma4<<<cdiv((1344 / 16) * (128 / 64), 8), 256, 0, stream>>>(
      pooled, 32, wupt, 32, b_up, tmpU, (_Float16*)nullptr, 1344, 128, 32, 0);
  h_assemble<<<cdiv(2 * S_TOT * 128, 256), 256, 0, stream>>>(x, tmpU, hf);
  ln_kernel<<<ROWS, 32, 0, stream>>>(hf, (const float*)nullptr, ln0_g, ln0_b,
                                     1e-5f, hf, hf16);

  // ---- transformer layers
  const int gT = cdiv((ROWS / 16) * (128 / 64), 8);     // 128-wide GEMMs (16x64 blocks)
  const int gF = cdiv((ROWS / 16) * (512 / 64), 8);     // 512-wide GEMM
  for (int l = 0; l < 2; ++l) {
    transpose16<<<cdiv(128 * 128, 256), 256, 0, stream>>>(wq + l * 16384, wqt, 128, 128);
    transpose16<<<cdiv(128 * 128, 256), 256, 0, stream>>>(wk + l * 16384, wkt, 128, 128);
    transpose16<<<cdiv(128 * 128, 256), 256, 0, stream>>>(wv + l * 16384, wvt, 128, 128);
    transpose16<<<cdiv(128 * 128, 256), 256, 0, stream>>>(wo + l * 16384, wot, 128, 128);
    transpose16<<<cdiv(128 * 512, 256), 256, 0, stream>>>(ffn_w1 + l * 128 * 512, w1t, 128, 512);
    transpose16<<<cdiv(512 * 128, 256), 256, 0, stream>>>(ffn_w2 + l * 512 * 128, w2t, 512, 128);

    gemm_wmma4<<<gT, 256, 0, stream>>>(hf16, 128, wqt, 128, (const float*)nullptr,
                                       (float*)nullptr, qf16, ROWS, 128, 128, 0);
    gemm_wmma4<<<gT, 256, 0, stream>>>(hf16, 128, wkt, 128, (const float*)nullptr,
                                       (float*)nullptr, kf16, ROWS, 128, 128, 0);
    gemm_wmma4<<<gT, 256, 0, stream>>>(hf16, 128, wvt, 128, (const float*)nullptr,
                                       (float*)nullptr, vf16, ROWS, 128, 128, 0);
    vtrans<<<cdiv(2 * S_TOT * 128, 256), 256, 0, stream>>>(vf16, vtf16);

    attn_kernel<<<2 * 4 * (S_TOT / 16), 32, 0, stream>>>(qf16, kf16, vtf16, mask, of16);

    gemm_wmma4<<<gT, 256, 0, stream>>>(of16, 128, wot, 128, (const float*)nullptr,
                                       tmp, (_Float16*)nullptr, ROWS, 128, 128, 0);
    ln_kernel<<<ROWS, 32, 0, stream>>>(hf, tmp, ln1_g + l * 128, ln1_b + l * 128,
                                       1e-6f, hf, hf16);

    gemm_wmma4<<<gF, 256, 0, stream>>>(hf16, 128, w1t, 128, ffn_b1 + l * 512,
                                       (float*)nullptr, f1f16, ROWS, 512, 128, 1);
    gemm_wmma4<<<gT, 256, 0, stream>>>(f1f16, 512, w2t, 512, ffn_b2 + l * 128,
                                       tmp, (_Float16*)nullptr, ROWS, 128, 512, 0);
    ln_kernel<<<ROWS, 32, 0, stream>>>(hf, tmp, ln2_g + l * 128, ln2_b + l * 128,
                                       1e-6f, hf, hf16);
  }

  // ---- final gather
  gather_out<<<cdiv(2 * 2048 * 4 * 128, 256), 256, 0, stream>>>(hf, indexes, out);
}
